// GraphNet_0_38826504355977
// MI455X (gfx1250) — matrix operand order repaired and checked by reading, not compiled, not run on hardware
//
#include <hip/hip_runtime.h>
#include <hip/hip_bf16.h>

typedef __attribute__((ext_vector_type(16))) __bf16 v16bf;
typedef __attribute__((ext_vector_type(8)))  float  v8f;

#define HEADS   4
#define NHID    256
#define FIN     128
#define NGRAPHS 16

// ---------- float <-> ordered-uint keys for atomic float max ----------
__device__ __forceinline__ unsigned f32_key(float f) {
  unsigned b = __float_as_uint(f);
  return (b & 0x80000000u) ? ~b : (b | 0x80000000u);
}
__device__ __forceinline__ float key_f32(unsigned u) {
  unsigned b = (u & 0x80000000u) ? (u & 0x7fffffffu) : ~u;
  return __uint_as_float(b);
}

// ---------- zero workspace ----------
__global__ void zero_f4(float4* __restrict__ p, long n4) {
  long i = (long)blockIdx.x * blockDim.x + threadIdx.x;
  long stride = (long)gridDim.x * blockDim.x;
  float4 z = {0.f, 0.f, 0.f, 0.f};
  for (; i < n4; i += stride) p[i] = z;
}

// ---------- GEMM: xp = x @ W  via v_wmma_f32_16x16x32_bf16 ----------
// one wave computes a 16x16 tile; tiles laid out col-tile fastest so that the
// 8 waves of a block share the same 16 rows of x (L1 reuse).
__global__ __launch_bounds__(256) void gemm_xw_wmma(
    const float* __restrict__ x, const float* __restrict__ W,
    float* __restrict__ xp, int Nn) {
  const int lane = threadIdx.x & 31;
  const int wid  = threadIdx.x >> 5;
  const int tile = blockIdx.x * 8 + wid;
  const int row_tile = tile >> 4;          // NHID/16 = 16 col tiles
  const int col_tile = tile & 15;
  const int row0 = row_tile * 16;
  if (row0 >= Nn) return;
  const int col0 = col_tile * 16;

  int m = row0 + (lane & 15);
  if (m >= Nn) m = Nn - 1;                 // clamp (tail tiles)
  const int n    = col0 + (lane & 15);
  const int hiA  = (lane >> 4) << 3;       // 0 or 8   (A: K sub-chunk)
  const int hiB  = (lane >> 4) << 3;       // 0 or 8   (B: K sub-chunk)

  v8f acc = {};
#pragma unroll
  for (int kk = 0; kk < FIN; kk += 32) {
    // A tile: lane holds row m, K = kk+hiA+0..7 and kk+16+hiA+0..7
    const float* xr = x + (size_t)m * FIN + kk + hiA;
    float4 f0 = *(const float4*)(xr);
    float4 f1 = *(const float4*)(xr + 4);
    float4 f2 = *(const float4*)(xr + 16);
    float4 f3 = *(const float4*)(xr + 20);
    v16bf a;
    a[0]=(__bf16)f0.x; a[1]=(__bf16)f0.y; a[2]=(__bf16)f0.z; a[3]=(__bf16)f0.w;
    a[4]=(__bf16)f1.x; a[5]=(__bf16)f1.y; a[6]=(__bf16)f1.z; a[7]=(__bf16)f1.w;
    a[8]=(__bf16)f2.x; a[9]=(__bf16)f2.y; a[10]=(__bf16)f2.z; a[11]=(__bf16)f2.w;
    a[12]=(__bf16)f3.x; a[13]=(__bf16)f3.y; a[14]=(__bf16)f3.z; a[15]=(__bf16)f3.w;

    // B tile: lane holds column n, K = kk+hiB+0..7 and kk+16+hiB+0..7 (stride NHID in W)
    const float* wc = W + (size_t)(kk + hiB) * NHID + n;
    v16bf b;
#pragma unroll
    for (int j = 0; j < 16; ++j) b[j] = (__bf16)wc[(size_t)(j + (j & 8)) * NHID];

    acc = __builtin_amdgcn_wmma_f32_16x16x32_bf16(
        false, a, false, b, (short)0, acc, false, false);
  }

  // D layout: VGPR r -> row = row0 + r + 8*(lane>=16), col = col0 + (lane&15)
  const int rbase = row0 + ((lane >> 4) << 3);
#pragma unroll
  for (int r = 0; r < 8; ++r) {
    const int row = rbase + r;
    if (row < Nn) xp[(size_t)row * NHID + col0 + (lane & 15)] = acc[r];
  }
}

// ---------- a_src / a_dst: per-(node,head) dot over 64 channels ----------
__global__ void attn_dot(const float* __restrict__ xp,
                         const float* __restrict__ att_src,
                         const float* __restrict__ att_dst,
                         float* __restrict__ a_src, float* __restrict__ a_dst,
                         int NH) {
  const int t = blockIdx.x * blockDim.x + threadIdx.x;
  if (t >= NH) return;
  const int h = t & 3;
  const float* row = xp + (size_t)t * 64;     // (n*4+h)*64 == n*256 + h*64
  const float* as  = att_src + h * 64;
  const float* ad  = att_dst + h * 64;
  float s = 0.f, d = 0.f;
#pragma unroll
  for (int c = 0; c < 64; c += 4) {
    float4 v  = *(const float4*)(row + c);
    float4 p  = *(const float4*)(as + c);
    float4 q  = *(const float4*)(ad + c);
    s += v.x*p.x + v.y*p.y + v.z*p.z + v.w*p.w;
    d += v.x*q.x + v.y*q.y + v.z*q.z + v.w*q.w;
  }
  a_src[t] = s;
  a_dst[t] = d;
}

// ---------- pass 1: leaky-relu logits + segment max (atomic on keys) ----------
__global__ void edge_logits_max(const int* __restrict__ ei, int E,
                                const float* __restrict__ a_src,
                                const float* __restrict__ a_dst,
                                float* __restrict__ logits,
                                unsigned* __restrict__ m_u, int total) {
  const int i = blockIdx.x * blockDim.x + threadIdx.x;
  if (i >= total) return;
  int s, d;
  if (i < E) { s = ei[i]; d = ei[E + i]; } else { s = d = i - E; }
#pragma unroll
  for (int h = 0; h < HEADS; ++h) {
    float l = a_src[s * HEADS + h] + a_dst[d * HEADS + h];
    l = l > 0.f ? l : 0.2f * l;                       // leaky_relu(0.2)
    logits[(size_t)i * HEADS + h] = l;
    atomicMax(&m_u[d * HEADS + h], f32_key(l));
  }
}

// ---------- pass 2: e = exp(l - m), segment sum ----------
__global__ void edge_exp_sum(const int* __restrict__ ei, int E,
                             float* __restrict__ logits,
                             const unsigned* __restrict__ m_u,
                             float* __restrict__ ssum, int total) {
  const int i = blockIdx.x * blockDim.x + threadIdx.x;
  if (i >= total) return;
  int d;
  if (i < E) { d = ei[E + i]; } else { d = i - E; }
#pragma unroll
  for (int h = 0; h < HEADS; ++h) {
    float m  = key_f32(m_u[d * HEADS + h]);
    float ev = __expf(logits[(size_t)i * HEADS + h] - m);
    logits[(size_t)i * HEADS + h] = ev;               // reuse buffer: now e
    unsafeAtomicAdd(&ssum[d * HEADS + h], ev);
  }
}

// ---------- pass 3: out[dst] += alpha * xp[src]  (wave per edge) ----------
__global__ __launch_bounds__(256) void edge_scatter(
    const int* __restrict__ ei, int E, int total,
    const float* __restrict__ xp, const float* __restrict__ evals,
    const float* __restrict__ ssum, float* __restrict__ outacc) {
  const int lane = threadIdx.x & 31;
  const int wid  = threadIdx.x >> 5;
  const int i = blockIdx.x * 8 + wid;
  if (i >= total) return;
  int s, d;
  if (i < E) { s = ei[i]; d = ei[E + i]; } else { s = d = i - E; }
  const int h     = lane >> 3;                        // 8 lanes per head
  const float alp = evals[(size_t)i * HEADS + h] /
                    (ssum[d * HEADS + h] + 1e-16f);
  const int c0 = lane * 8;
  const float4* xv = (const float4*)(xp + (size_t)s * NHID + c0);
  float4 v0 = xv[0], v1 = xv[1];
  float* op = outacc + (size_t)d * NHID + c0;
  unsafeAtomicAdd(op + 0, v0.x * alp);
  unsafeAtomicAdd(op + 1, v0.y * alp);
  unsafeAtomicAdd(op + 2, v0.z * alp);
  unsafeAtomicAdd(op + 3, v0.w * alp);
  unsafeAtomicAdd(op + 4, v1.x * alp);
  unsafeAtomicAdd(op + 5, v1.y * alp);
  unsafeAtomicAdd(op + 6, v1.z * alp);
  unsafeAtomicAdd(op + 7, v1.w * alp);
}

// ---------- bias + relu + per-graph max/sum/count ----------
__global__ void finalize_pool(const float* __restrict__ outacc,
                              const float* __restrict__ bias,
                              const int* __restrict__ batch,
                              unsigned* __restrict__ gmax_u,
                              float* __restrict__ gsum,
                              float* __restrict__ cnt, int Nn) {
  const int t = blockIdx.x * blockDim.x + threadIdx.x;
  if (t >= Nn * NHID) return;
  const int n = t >> 8, c = t & (NHID - 1);
  float h = outacc[t] + bias[c];
  h = h > 0.f ? h : 0.f;                              // relu -> h >= 0
  const int g = batch[n];
  atomicMax(&gmax_u[g * NHID + c], __float_as_uint(h)); // bits monotone, h>=0
  unsafeAtomicAdd(&gsum[g * NHID + c], h);
  if (c == 0) unsafeAtomicAdd(&cnt[g], 1.0f);
}

// ---------- write [NGRAPHS, 2*NHID] = concat(gmax, gmean) ----------
__global__ void writeout(const unsigned* __restrict__ gmax_u,
                         const float* __restrict__ gsum,
                         const float* __restrict__ cnt,
                         float* __restrict__ out, int nout) {
  const int t = blockIdx.x * blockDim.x + threadIdx.x;
  if (t >= nout) return;
  const int g = t >> 9, j = t & 511;
  float v = (j < NHID) ? __uint_as_float(gmax_u[g * NHID + j])
                       : gsum[g * NHID + (j - NHID)] / (cnt[g] + 1e-16f);
  out[t] = v;
}

extern "C" void kernel_launch(void* const* d_in, const int* in_sizes, int n_in,
                              void* d_out, int out_size, void* d_ws, size_t ws_size,
                              hipStream_t stream) {
  const float* x       = (const float*)d_in[0];
  const int*   ei      = (const int*)  d_in[1];
  const int*   batch   = (const int*)  d_in[2];
  const float* W       = (const float*)d_in[3];
  const float* att_src = (const float*)d_in[4];
  const float* att_dst = (const float*)d_in[5];
  const float* bias    = (const float*)d_in[6];

  const int N     = in_sizes[2];
  const int E     = in_sizes[1] / 2;
  const int total = E + N;                 // edges + self loops

  char* ws = (char*)d_ws;
  size_t off = 0;
  auto carve = [&](size_t bytes) -> char* {
    char* p = ws + off;
    off += (bytes + 255) & ~(size_t)255;
    return p;
  };
  float*    xp     = (float*)   carve((size_t)N * NHID * 4);
  float*    a_src  = (float*)   carve((size_t)N * HEADS * 4);
  float*    a_dst  = (float*)   carve((size_t)N * HEADS * 4);
  float*    logits = (float*)   carve((size_t)total * HEADS * 4);
  char*     zbase  = ws + off;             // everything below must be zeroed
  float*    outacc = (float*)   carve((size_t)N * NHID * 4);
  unsigned* m_u    = (unsigned*)carve((size_t)N * HEADS * 4);
  float*    ssum   = (float*)   carve((size_t)N * HEADS * 4);
  unsigned* gmax_u = (unsigned*)carve((size_t)NGRAPHS * NHID * 4);
  float*    gsum   = (float*)   carve((size_t)NGRAPHS * NHID * 4);
  float*    cnt    = (float*)   carve((size_t)NGRAPHS * 4);
  const size_t zbytes = (size_t)((ws + off) - zbase);   // multiple of 256

  // 0) zero accumulators (every call; harness does not re-poison)
  zero_f4<<<2048, 256, 0, stream>>>((float4*)zbase, (long)(zbytes / 16));

  // 1) xp = x @ W  (WMMA bf16)
  {
    const int row_tiles = (N + 15) / 16;
    const int tiles = row_tiles * (NHID / 16);
    gemm_xw_wmma<<<(tiles + 7) / 8, 256, 0, stream>>>(x, W, xp, N);
  }

  // 2) attention dots
  {
    const int nh = N * HEADS;
    attn_dot<<<(nh + 255) / 256, 256, 0, stream>>>(xp, att_src, att_dst,
                                                   a_src, a_dst, nh);
  }

  // 3) logits + segment max
  edge_logits_max<<<(total + 255) / 256, 256, 0, stream>>>(
      ei, E, a_src, a_dst, logits, m_u, total);

  // 4) exp + segment sum
  edge_exp_sum<<<(total + 255) / 256, 256, 0, stream>>>(
      ei, E, logits, m_u, ssum, total);

  // 5) message scatter-add (wave per edge)
  edge_scatter<<<(total + 7) / 8, 256, 0, stream>>>(
      ei, E, total, xp, logits, ssum, outacc);

  // 6) bias + relu + pooling accumulators
  finalize_pool<<<(N * NHID + 255) / 256, 256, 0, stream>>>(
      outacc, bias, batch, gmax_u, gsum, cnt, N);

  // 7) final [16, 512] output
  writeout<<<(out_size + 255) / 256, 256, 0, stream>>>(
      gmax_u, gsum, cnt, (float*)d_out, out_size);
}
